// ImageCaptioningModel_63144609185924
// MI455X (gfx1250) — matrix operand order repaired and checked by reading, not compiled, not run on hardware
//
#include <hip/hip_runtime.h>
#include <math.h>

// ---------------- types for CDNA5 WMMA ----------------
typedef __bf16 bf16;
typedef __attribute__((ext_vector_type(16))) __bf16 v16bf;
typedef __attribute__((ext_vector_type(8)))  float  v8f;

// ---------------- problem dims ----------------
#define Bsz   64
#define Pn    49
#define Tn    30
#define ENCn  2048
#define DECn  512
#define ATTn  512
#define EMBn  256
#define VOCn  10000

// =====================================================================
// WMMA tile operand loaders (wave32, 16x16x32 bf16 layouts per CDNA5 ISA)
//  A (16x32, bf16): lane half h=lane>>4; VGPR v<4 -> K = 8h+2v(+1),
//                   v>=4 -> K = 16+8h+2(v-4)(+1)  => two contiguous 16B runs.
//  B (32x16, bf16): same K mapping as A over W[N][K] rows => two contiguous 16B runs.
// =====================================================================
__device__ __forceinline__ v16bf load_a_bf16(const bf16* A, int lda, int row, int kt, int kh) {
  union { v16bf v; uint4 q[2]; } x;
  const bf16* p = A + (size_t)row * lda + kt + kh * 8;
  x.q[0] = *(const uint4*)p;
  x.q[1] = *(const uint4*)(p + 16);
  return x.v;
}
__device__ __forceinline__ v16bf load_b_bf16(const bf16* W, int ldw, int col, int kt, int kh) {
  union { v16bf v; uint4 q[2]; } x;
  const bf16* p = W + (size_t)col * ldw + kt + kh * 8;
  x.q[0] = *(const uint4*)p;
  x.q[1] = *(const uint4*)(p + 16);
  return x.v;
}

// =====================================================================
// Generic GEMM:  C[M,N] = A1[M,K1] * W1[N,K1]^T (+ A2[M,K2] * W2[N,K2]^T)
//                (+ bias1[N]) (+ bias2[N])
// Register-blocked MBLK row-tiles per wave. Per K-step: issue B + MBLK A
// tile loads as one clause, then MBLK back-to-back v_wmma_f32_16x16x32_bf16.
// =====================================================================
template <int MBLK>
__global__ void wmma_gemm_kernel(const bf16* A1, int lda1, int K1, const bf16* Wt1,
                                 const bf16* A2, int lda2, int K2, const bf16* Wt2,
                                 const float* bias1, const float* bias2,
                                 float* C, int ldc, int M, int N) {
  const int wid     = blockIdx.x * (blockDim.x >> 5) + (threadIdx.x >> 5);
  const int tilesN  = N >> 4;
  const int groupsM = M / (16 * MBLK);
  const int tiles   = groupsM * tilesN;
  if (wid >= tiles) return;                 // whole-wave uniform exit (EXEC stays all-ones)
  const int tg   = wid / tilesN;
  const int tn   = wid - tg * tilesN;
  const int lane = threadIdx.x & 31;
  const int kh   = lane >> 4;
  const int l15  = lane & 15;
  const int row0 = tg * (16 * MBLK) + l15;
  const int col  = tn * 16 + l15;

  v8f acc[MBLK];
#pragma unroll
  for (int mb = 0; mb < MBLK; ++mb) acc[mb] = (v8f){0.f,0.f,0.f,0.f,0.f,0.f,0.f,0.f};

#pragma unroll 2
  for (int kt = 0; kt < K1; kt += 32) {
    v16bf b = load_b_bf16(Wt1, K1, col, kt, kh);
    v16bf a[MBLK];
#pragma unroll
    for (int mb = 0; mb < MBLK; ++mb) a[mb] = load_a_bf16(A1, lda1, row0 + mb * 16, kt, kh);
#pragma unroll
    for (int mb = 0; mb < MBLK; ++mb)
      acc[mb] = __builtin_amdgcn_wmma_f32_16x16x32_bf16(false, a[mb], false, b, (short)0,
                                                        acc[mb], false, false);
  }
  if (A2) {
#pragma unroll 2
    for (int kt = 0; kt < K2; kt += 32) {
      v16bf b = load_b_bf16(Wt2, K2, col, kt, kh);
      v16bf a[MBLK];
#pragma unroll
      for (int mb = 0; mb < MBLK; ++mb) a[mb] = load_a_bf16(A2, lda2, row0 + mb * 16, kt, kh);
#pragma unroll
      for (int mb = 0; mb < MBLK; ++mb)
        acc[mb] = __builtin_amdgcn_wmma_f32_16x16x32_bf16(false, a[mb], false, b, (short)0,
                                                          acc[mb], false, false);
    }
  }
  float bsum = 0.f;
  if (bias1) bsum += bias1[col];
  if (bias2) bsum += bias2[col];
#pragma unroll
  for (int mb = 0; mb < MBLK; ++mb) {
#pragma unroll
    for (int r = 0; r < 8; ++r) {
      const int m = tg * (16 * MBLK) + mb * 16 + r + 8 * kh;   // C/D: VGPR r, lane-half kh
      C[(size_t)m * ldc + col] = acc[mb][r] + bsum;
    }
  }
}

// ---------------- fp32 -> bf16 convert ----------------
__global__ void cvt_bf16_kernel(const float* __restrict__ src, bf16* __restrict__ dst, int n) {
  int i = blockIdx.x * blockDim.x + threadIdx.x;
  if (i < n) dst[i] = (bf16)src[i];
}

// ---------------- embedding gather (captions -> bf16 embeddings) ----------------
__global__ void emb_gather_kernel(const float* __restrict__ emb, const int* __restrict__ captions,
                                  bf16* __restrict__ out) {
  int i = blockIdx.x * blockDim.x + threadIdx.x;      // over B*T*EMB
  if (i >= Bsz * Tn * EMBn) return;
  int e  = i & (EMBn - 1);
  int bt = i >> 8;                                    // EMBn == 256
  int tok = captions[bt];
  out[i] = (bf16)emb[(size_t)tok * EMBn + e];
}

// ---------------- zero-init LSTM state ----------------
__global__ void init_state_kernel(float* c1, float* c2, bf16* h1, bf16* h2) {
  int i = blockIdx.x * blockDim.x + threadIdx.x;      // over B*DEC
  if (i < Bsz * DECn) {
    c1[i] = 0.f; c2[i] = 0.f;
    h1[i] = (bf16)0.f; h2[i] = (bf16)0.f;
  }
}

// =====================================================================
// Attention: per block b:
//   s_p = bV + sum_a V[a]*tanh(featW1[b,p,a] + h2W2[b,a])   (b1,b2 folded in)
//   w = softmax_p(s);  ctx[e] = sum_p w_p * features[b,p,e]
//   lstm_in[b] = concat(emb[caption_t], ctx) as bf16
// =====================================================================
__global__ void attention_kernel(const float* __restrict__ featW1, const float* __restrict__ h2w2,
                                 const float* __restrict__ Vw, const float* __restrict__ bV,
                                 const float* __restrict__ features,
                                 const bf16* __restrict__ embs, int t,
                                 bf16* __restrict__ lstm_in) {
  __shared__ float sc[64];
  __shared__ float wgt[64];
  const int b    = blockIdx.x;
  const int tid  = threadIdx.x;
  const int wave = tid >> 5;
  const int lane = tid & 31;
  const float* fw = featW1 + (size_t)b * Pn * ATTn;
  const float* hw = h2w2   + (size_t)b * ATTn;

  for (int p = wave; p < Pn; p += 8) {
    float s = 0.f;
    for (int a = lane; a < ATTn; a += 32)
      s += Vw[a] * tanhf(fw[(size_t)p * ATTn + a] + hw[a]);
#pragma unroll
    for (int off = 16; off > 0; off >>= 1) s += __shfl_xor(s, off, 32);
    if (lane == 0) sc[p] = s + bV[0];
  }
  __syncthreads();

  if (tid < Pn) {   // redundant per-thread softmax stats (49 elems, trivially cheap)
    float m = -1e30f;
    for (int p = 0; p < Pn; ++p) m = fmaxf(m, sc[p]);
    float den = 0.f;
    for (int p = 0; p < Pn; ++p) den += expf(sc[p] - m);
    wgt[tid] = expf(sc[tid] - m) / den;
  }
  __syncthreads();

  const float* feat = features + (size_t)b * Pn * ENCn;
  bf16* lin = lstm_in + (size_t)b * (EMBn + ENCn);
  for (int e = tid; e < ENCn; e += blockDim.x) {
    float acc = 0.f;
#pragma unroll 7
    for (int p = 0; p < Pn; ++p) acc += wgt[p] * feat[(size_t)p * ENCn + e];
    lin[EMBn + e] = (bf16)acc;
  }
  const bf16* xe = embs + ((size_t)b * Tn + t) * EMBn;
  for (int e = tid; e < EMBn; e += blockDim.x) lin[e] = xe[e];
}

// ---------------- LSTM cell (PyTorch gate order i,f,g,o), fp32 state ----------------
__global__ void lstm_cell_kernel(const float* __restrict__ g, float* __restrict__ c,
                                 bf16* __restrict__ h) {
  int i = blockIdx.x * blockDim.x + threadIdx.x;      // over B*DEC
  if (i >= Bsz * DECn) return;
  int b = i >> 9, d = i & (DECn - 1);                 // DECn == 512
  const float* gb = g + (size_t)b * 4 * DECn;
  float gi = gb[d], gf = gb[DECn + d], gg = gb[2 * DECn + d], go = gb[3 * DECn + d];
  float si = 1.f / (1.f + expf(-gi));
  float sf = 1.f / (1.f + expf(-gf));
  float so = 1.f / (1.f + expf(-go));
  float cn = sf * c[i] + si * tanhf(gg);
  c[i] = cn;
  h[i] = (bf16)(so * tanhf(cn));
}

// =====================================================================
// Host launcher
// =====================================================================
extern "C" void kernel_launch(void* const* d_in, const int* in_sizes, int n_in,
                              void* d_out, int out_size, void* d_ws, size_t ws_size,
                              hipStream_t stream) {
  (void)in_sizes; (void)n_in; (void)out_size; (void)ws_size;
  const float* features = (const float*)d_in[0];
  const int*   captions = (const int*)d_in[1];
  const float* emb      = (const float*)d_in[2];
  const float* W1    = (const float*)d_in[3];
  const float* b1    = (const float*)d_in[4];
  const float* W2    = (const float*)d_in[5];
  const float* b2    = (const float*)d_in[6];
  const float* Vw    = (const float*)d_in[7];
  const float* bV    = (const float*)d_in[8];
  const float* W_ih1 = (const float*)d_in[9];
  const float* W_hh1 = (const float*)d_in[10];
  const float* b_ih1 = (const float*)d_in[11];
  const float* b_hh1 = (const float*)d_in[12];
  const float* W_ih2 = (const float*)d_in[13];
  const float* W_hh2 = (const float*)d_in[14];
  const float* b_ih2 = (const float*)d_in[15];
  const float* b_hh2 = (const float*)d_in[16];
  const float* fc_W  = (const float*)d_in[17];
  const float* fc_b  = (const float*)d_in[18];
  float* out = (float*)d_out;

  // ---- carve workspace (~51 MB) ----
  char* ws = (char*)d_ws;
  size_t off = 0;
  auto carve = [&](size_t bytes) -> void* {
    void* p = ws + off;
    off += (bytes + 255) & ~(size_t)255;
    return p;
  };
  bf16* W1b    = (bf16*)carve((size_t)ATTn * ENCn * 2);
  bf16* W2b    = (bf16*)carve((size_t)ATTn * DECn * 2);
  bf16* Wih1b  = (bf16*)carve((size_t)4 * DECn * (EMBn + ENCn) * 2);
  bf16* Whh1b  = (bf16*)carve((size_t)4 * DECn * DECn * 2);
  bf16* Wih2b  = (bf16*)carve((size_t)4 * DECn * DECn * 2);
  bf16* Whh2b  = (bf16*)carve((size_t)4 * DECn * DECn * 2);
  bf16* fcWb   = (bf16*)carve((size_t)VOCn * DECn * 2);
  bf16* featb  = (bf16*)carve((size_t)Bsz * Pn * ENCn * 2);   // bf16 features (GEMM A)
  bf16* embsb  = (bf16*)carve((size_t)Bsz * Tn * EMBn * 2);
  float* featW1 = (float*)carve((size_t)Bsz * Pn * ATTn * 4);
  float* h2w2   = (float*)carve((size_t)Bsz * ATTn * 4);
  float* g1     = (float*)carve((size_t)Bsz * 4 * DECn * 4);
  float* g2     = (float*)carve((size_t)Bsz * 4 * DECn * 4);
  float* c1     = (float*)carve((size_t)Bsz * DECn * 4);
  float* c2     = (float*)carve((size_t)Bsz * DECn * 4);
  bf16*  h1b    = (bf16*)carve((size_t)Bsz * DECn * 2);
  bf16*  h2b    = (bf16*)carve((size_t)Bsz * DECn * 2);
  bf16*  lstmin = (bf16*)carve((size_t)Bsz * (EMBn + ENCn) * 2);

  auto cvt = [&](const float* s, bf16* d, int n) {
    cvt_bf16_kernel<<<(n + 255) / 256, 256, 0, stream>>>(s, d, n);
  };
  auto gemm = [&](int mblk, const bf16* A1, int lda1, int K1, const bf16* Wt1,
                  const bf16* A2, int lda2, int K2, const bf16* Wt2,
                  const float* bias1, const float* bias2,
                  float* C, int ldc, int M, int N) {
    int tiles = (M / (16 * mblk)) * (N / 16);
    int grid  = (tiles + 7) / 8;               // 8 waves (256 threads) per block
    if (mblk == 4) {
      wmma_gemm_kernel<4><<<grid, 256, 0, stream>>>(A1, lda1, K1, Wt1, A2, lda2, K2, Wt2,
                                                    bias1, bias2, C, ldc, M, N);
    } else {
      wmma_gemm_kernel<1><<<grid, 256, 0, stream>>>(A1, lda1, K1, Wt1, A2, lda2, K2, Wt2,
                                                    bias1, bias2, C, ldc, M, N);
    }
  };

  // ---- one-time: convert weights + features to bf16 (L2-resident for all 30 steps) ----
  cvt(W1,    W1b,   ATTn * ENCn);
  cvt(W2,    W2b,   ATTn * DECn);
  cvt(W_ih1, Wih1b, 4 * DECn * (EMBn + ENCn));
  cvt(W_hh1, Whh1b, 4 * DECn * DECn);
  cvt(W_ih2, Wih2b, 4 * DECn * DECn);
  cvt(W_hh2, Whh2b, 4 * DECn * DECn);
  cvt(fc_W,  fcWb,  VOCn * DECn);
  cvt(features, featb, Bsz * Pn * ENCn);
  emb_gather_kernel<<<(Bsz * Tn * EMBn + 255) / 256, 256, 0, stream>>>(emb, captions, embsb);
  init_state_kernel<<<(Bsz * DECn + 255) / 256, 256, 0, stream>>>(c1, c2, h1b, h2b);

  // ---- one-time: featW1[B*P, ATT] = features[B*P, ENC] @ W1^T + b1  (1568 waves) ----
  gemm(4, featb, ENCn, ENCn, W1b, nullptr, 0, 0, nullptr,
       b1, nullptr, featW1, ATTn, Bsz * Pn, ATTn);

  // ---- 30 serial decode steps ----
  for (int t = 0; t < Tn; ++t) {
    // h2@W2^T + b2   (tiny, latency-critical -> MBLK=1 for 128-wave parallelism)
    gemm(1, h2b, DECn, DECn, W2b, nullptr, 0, 0, nullptr,
         b2, nullptr, h2w2, ATTn, Bsz, ATTn);
    // attention + build lstm_in = [emb_t, ctx]
    attention_kernel<<<Bsz, 256, 0, stream>>>(featW1, h2w2, Vw, bV, features, embsb, t, lstmin);
    // LSTM1 gates: lstm_in@W_ih1^T + h1@W_hh1^T + b_ih1 + b_hh1  (9.4MB weights -> MBLK=4)
    gemm(4, lstmin, EMBn + ENCn, EMBn + ENCn, Wih1b,
         h1b, DECn, DECn, Whh1b, b_ih1, b_hh1, g1, 4 * DECn, Bsz, 4 * DECn);
    lstm_cell_kernel<<<(Bsz * DECn + 255) / 256, 256, 0, stream>>>(g1, c1, h1b);
    // LSTM2 gates: h1@W_ih2^T + h2@W_hh2^T + b_ih2 + b_hh2
    gemm(4, h1b, DECn, DECn, Wih2b,
         h2b, DECn, DECn, Whh2b, b_ih2, b_hh2, g2, 4 * DECn, Bsz, 4 * DECn);
    lstm_cell_kernel<<<(Bsz * DECn + 255) / 256, 256, 0, stream>>>(g2, c2, h2b);
    // logits: out[:, t, :] = h2@fc_W^T + fc_b  (10MB weights, 625 waves at MBLK=4)
    gemm(4, h2b, DECn, DECn, fcWb, nullptr, 0, 0, nullptr,
         fc_b, nullptr, out + (size_t)t * VOCn, Tn * VOCn, Bsz, VOCn);
  }
}
